// SigmaMoE_37177236914769
// MI455X (gfx1250) — matrix operand order repaired and checked by reading, not compiled, not run on hardware
//
#include <hip/hip_runtime.h>
#include <math.h>

// ------------------------------------------------------------------
// SigmaMoE forward for MI455X (gfx1250): bf16 WMMA + TDM tile staging.
// ------------------------------------------------------------------

typedef __attribute__((ext_vector_type(16))) __bf16 v16bf;
typedef __attribute__((ext_vector_type(8)))  float  v8f;
typedef __attribute__((ext_vector_type(4)))  unsigned int u32x4;
typedef __attribute__((ext_vector_type(8)))  int          i32x8;
typedef __attribute__((ext_vector_type(4)))  int          i32x4;

union FragU { v16bf v; uint4 q[2]; };

#define T_DIM 4096
#define D_DIM 1024
#define I_DIM 512
#define E_DIM 8

#define BM 128
#define BN 128
#define BK 32
#define SK 40   // padded LDS row stride in ushorts (64B row + 16B pad)

#if __has_builtin(__builtin_amdgcn_tensor_load_to_lds) && \
    __has_builtin(__builtin_amdgcn_s_wait_tensorcnt)
#define USE_TDM 1
#else
#define USE_TDM 0
#endif

__device__ __forceinline__ unsigned short f2bf(float f) {
  unsigned int u = __float_as_uint(f);
  u += 0x7FFFu + ((u >> 16) & 1u);     // round-to-nearest-even
  return (unsigned short)(u >> 16);
}

// A fragment (16x32 bf16, M x K): lanes 0-15 row M=lane hold K={0..7,16..23},
// lanes 16-31 row M=lane-16 hold K={8..15,24..31}.  LDS tile is [m][k] row-major.
__device__ __forceinline__ v16bf ldsA_frag(const unsigned short* s, int m, int lane) {
  const int kh = (lane >> 4) * 8;
  FragU f;
  f.q[0] = *reinterpret_cast<const uint4*>(s + m * SK + kh);
  f.q[1] = *reinterpret_cast<const uint4*>(s + m * SK + 16 + kh);
  return f.v;
}

// B fragment (32x16 bf16, K x N): lanes 0-15 col N=lane hold K={0..7,16..23},
// lanes 16-31 col N=lane-16 hold K={8..15,24..31}.  LDS tile is [n][k] row-major.
__device__ __forceinline__ v16bf ldsB_frag(const unsigned short* s, int n, int lane) {
  const int kq = (lane >> 4) * 8;
  FragU f;
  f.q[0] = *reinterpret_cast<const uint4*>(s + n * SK + kq);
  f.q[1] = *reinterpret_cast<const uint4*>(s + n * SK + 16 + kq);
  return f.v;
}

#if USE_TDM
// One TDM DMA: load a BM(tile rows) x BK(contiguous bf16) tile into LDS with
// hardware padding of 4 DWORDs every 16 DWORDs  ->  LDS row stride = SK ushorts.
// D# layout per CDNA5 ISA 8.3/8.4 (group0: count/lds/global/type, group1: dims).
// This toolchain exposes the 6-arg builtin:
//   (u32x4 g0, i32x8 g1, i32x4 g2, i32x4 g3, i32x8 extra, i32 cpol)
__device__ __forceinline__ void tdm_load_tile(const unsigned short* gtile,
                                              unsigned int lds_off,
                                              int tensor_rows, int tensor_cols,
                                              int tile_rows) {
  const unsigned long long ga = (unsigned long long)(uintptr_t)gtile;
  u32x4 g0;
  g0[0] = 1u;                                              // count=1 (valid), user mode
  g0[1] = lds_off;                                         // lds_addr (bytes)
  g0[2] = (unsigned int)(ga & 0xFFFFFFFFull);              // global_addr[31:0]
  g0[3] = (unsigned int)((ga >> 32) & 0x01FFFFFFull)       // global_addr[56:32]
          | (2u << 30);                                    // type = 2 ("image")
  i32x8 g1;
  g1[0] = (1 << 16)                                        // data_size = 2 bytes
        | (1 << 20)                                        // pad_enable
        | (3 << 22)                                        // pad_interval: every 16 DW (64B)
        | (3 << 25);                                       // pad_amount: 4 DW (16B)
  g1[1] = (int)(((unsigned)tensor_cols & 0xFFFFu) << 16);  // tensor_dim0[15:0]
  g1[2] = (int)((((unsigned)tensor_cols >> 16) & 0xFFFFu)  // tensor_dim0[31:16]
        | (((unsigned)tensor_rows & 0xFFFFu) << 16));      // tensor_dim1[15:0]
  g1[3] = (int)((((unsigned)tensor_rows >> 16) & 0xFFFFu)  // tensor_dim1[31:16]
        | ((unsigned)BK << 16));                           // tile_dim0 = 32
  g1[4] = tile_rows;                                       // tile_dim1 (tile_dim2 = 0)
  g1[5] = tensor_cols;                                     // tensor_dim0_stride[31:0]
  g1[6] = 0;
  g1[7] = 0;
  const i32x4 gz4 = {0, 0, 0, 0};                          // 2-D tile: groups 2/3 unused
  const i32x8 gz8 = {0, 0, 0, 0, 0, 0, 0, 0};
  __builtin_amdgcn_tensor_load_to_lds(g0, g1, gz4, gz4, gz8, 0);
}
#endif

// ------------------------- router ---------------------------------
__global__ __launch_bounds__(256) void router_kernel(
    const float* __restrict__ x, const float* __restrict__ gate_w,
    float* __restrict__ combine) {
  const int lane = threadIdx.x & 31;
  const int t = blockIdx.x * 8 + (threadIdx.x >> 5);
  if (t >= T_DIM) return;

  float p[E_DIM];
#pragma unroll
  for (int e = 0; e < E_DIM; e++) p[e] = 0.0f;

  for (int k = lane; k < D_DIM; k += 32) {
    const float xv = x[(size_t)t * D_DIM + k];
#pragma unroll
    for (int e = 0; e < E_DIM; e++) p[e] += xv * gate_w[(size_t)e * D_DIM + k];
  }
#pragma unroll
  for (int e = 0; e < E_DIM; e++)
    for (int off = 16; off > 0; off >>= 1) p[e] += __shfl_xor(p[e], off, 32);

  if (lane == 0) {
    float mx = p[0];
#pragma unroll
    for (int e = 1; e < E_DIM; e++) mx = fmaxf(mx, p[e]);
    float s[E_DIM], sum = 0.0f;
#pragma unroll
    for (int e = 0; e < E_DIM; e++) { s[e] = expf(p[e] - mx); sum += s[e]; }
#pragma unroll
    for (int e = 0; e < E_DIM; e++) s[e] /= sum;
    int i1 = 0;
#pragma unroll
    for (int e = 1; e < E_DIM; e++) if (s[e] > s[i1]) i1 = e;
    int i2 = (i1 == 0) ? 1 : 0;
#pragma unroll
    for (int e = 0; e < E_DIM; e++) if (e != i1 && s[e] > s[i2]) i2 = e;
    const float norm = s[i1] + s[i2] + 1e-20f;
#pragma unroll
    for (int e = 0; e < E_DIM; e++)
      combine[(size_t)t * E_DIM + e] =
          (e == i1) ? s[i1] / norm : ((e == i2) ? s[i2] / norm : 0.0f);
  }
}

// ---------------------- fp32 -> bf16 converts ---------------------
__global__ void cvt_kernel(const float* __restrict__ in,
                           unsigned short* __restrict__ out, int n) {
  for (int i = blockIdx.x * blockDim.x + threadIdx.x; i < n;
       i += gridDim.x * blockDim.x)
    out[i] = f2bf(in[i]);
}

// in: [K][N] fp32 row-major -> out: [N][K] bf16 row-major (transposed)
__global__ void cvtT_kernel(const float* __restrict__ in,
                            unsigned short* __restrict__ out, int K, int N) {
  const int total = K * N;
  for (int i = blockIdx.x * blockDim.x + threadIdx.x; i < total;
       i += gridDim.x * blockDim.x) {
    const int k = i / N;
    const int n = i - k * N;
    out[(size_t)n * K + k] = f2bf(in[i]);
  }
}

// -------------------- per-tile compute helpers --------------------
__device__ __forceinline__ void gemm1_tile(const unsigned short* sA,
                                           const unsigned short* sU,
                                           const unsigned short* sG,
                                           int wm, int wn, int lane,
                                           v8f accU[2][4], v8f accG[2][4]) {
  v16bf afr[2];
  afr[0] = ldsA_frag(sA, wm * 32 + (lane & 15), lane);
  afr[1] = ldsA_frag(sA, wm * 32 + 16 + (lane & 15), lane);
#pragma unroll
  for (int nf = 0; nf < 4; nf++) {
    const int ncol = wn * 64 + nf * 16 + (lane & 15);
    v16bf bu = ldsB_frag(sU, ncol, lane);
    v16bf bg = ldsB_frag(sG, ncol, lane);
#pragma unroll
    for (int mf = 0; mf < 2; mf++) {
      accU[mf][nf] = __builtin_amdgcn_wmma_f32_16x16x32_bf16(
          false, afr[mf], false, bu, (short)0, accU[mf][nf], false, false);
      accG[mf][nf] = __builtin_amdgcn_wmma_f32_16x16x32_bf16(
          false, afr[mf], false, bg, (short)0, accG[mf][nf], false, false);
    }
  }
}

__device__ __forceinline__ void gemm2_tile(const unsigned short* sA,
                                           const unsigned short* sB,
                                           int wm, int wn, int lane,
                                           v8f acc[2][4]) {
  v16bf afr[2];
  afr[0] = ldsA_frag(sA, wm * 32 + (lane & 15), lane);
  afr[1] = ldsA_frag(sA, wm * 32 + 16 + (lane & 15), lane);
#pragma unroll
  for (int nf = 0; nf < 4; nf++) {
    const int ncol = wn * 64 + nf * 16 + (lane & 15);
    v16bf bf = ldsB_frag(sB, ncol, lane);
#pragma unroll
    for (int mf = 0; mf < 2; mf++) {
      acc[mf][nf] = __builtin_amdgcn_wmma_f32_16x16x32_bf16(
          false, afr[mf], false, bf, (short)0, acc[mf][nf], false, false);
    }
  }
}

// -------------------- GEMM1: H = up(x) * gelu(gate(x)) ------------
// x_bf: [T][D] bf16   upT/gateT: [I][D] bf16   H: [T][I] bf16
__global__ __launch_bounds__(256) void gemm1_upgate(
    const unsigned short* __restrict__ xbf,
    const unsigned short* __restrict__ upT,
    const unsigned short* __restrict__ gateT,
    unsigned short* __restrict__ H) {
  __shared__ unsigned short sA[2][BM * SK];
  __shared__ unsigned short sU[2][BN * SK];
  __shared__ unsigned short sG[2][BN * SK];

  const int tid  = threadIdx.x;
  const int lane = tid & 31;
  const int wid  = tid >> 5;
  const int wm   = wid & 3;    // 4 wave-rows  of 32
  const int wn   = wid >> 2;   // 2 wave-cols  of 64
  const int m0   = blockIdx.x * BM;
  const int n0   = blockIdx.y * BN;

  v8f accU[2][4], accG[2][4];
  const v8f vz = {0.f, 0.f, 0.f, 0.f, 0.f, 0.f, 0.f, 0.f};
#pragma unroll
  for (int i = 0; i < 2; i++)
#pragma unroll
    for (int j = 0; j < 4; j++) { accU[i][j] = vz; accG[i][j] = vz; }

  const int NK = D_DIM / BK;

#if USE_TDM
  unsigned int oA[2], oU[2], oG[2];
#pragma unroll
  for (int b = 0; b < 2; b++) {
    oA[b] = (unsigned int)(uintptr_t)(const void*)&sA[b][0];
    oU[b] = (unsigned int)(uintptr_t)(const void*)&sU[b][0];
    oG[b] = (unsigned int)(uintptr_t)(const void*)&sG[b][0];
  }
  // prologue: stage tile 0 into buffer 0 (one DMA per tensor, waves 0..2)
  if (wid == 0)      tdm_load_tile(xbf   + (size_t)m0 * D_DIM, oA[0], T_DIM, D_DIM, BM);
  else if (wid == 1) tdm_load_tile(upT   + (size_t)n0 * D_DIM, oU[0], I_DIM, D_DIM, BN);
  else if (wid == 2) tdm_load_tile(gateT + (size_t)n0 * D_DIM, oG[0], I_DIM, D_DIM, BN);

  for (int kt = 0; kt < NK; kt++) {
    const int cur = kt & 1, nxt = cur ^ 1;
    if (kt + 1 < NK) {
      const int k1 = (kt + 1) * BK;
      if (wid == 0)      tdm_load_tile(xbf   + (size_t)m0 * D_DIM + k1, oA[nxt], T_DIM, D_DIM, BM);
      else if (wid == 1) tdm_load_tile(upT   + (size_t)n0 * D_DIM + k1, oU[nxt], I_DIM, D_DIM, BN);
      else if (wid == 2) tdm_load_tile(gateT + (size_t)n0 * D_DIM + k1, oG[nxt], I_DIM, D_DIM, BN);
      __builtin_amdgcn_s_wait_tensorcnt(1);   // tile kt landed; kt+1 in flight
    } else {
      __builtin_amdgcn_s_wait_tensorcnt(0);
    }
    __syncthreads();
    gemm1_tile(sA[cur], sU[cur], sG[cur], wm, wn, lane, accU, accG);
    __syncthreads();
  }
#else
  const int lrow = tid >> 1;
  const int lseg = (tid & 1) * 16;
  for (int kt = 0; kt < NK; kt++) {
    const int k0 = kt * BK;
    {
      const uint4* ga = reinterpret_cast<const uint4*>(
          xbf + (size_t)(m0 + lrow) * D_DIM + k0 + lseg);
      *reinterpret_cast<uint4*>(sA[0] + lrow * SK + lseg)     = ga[0];
      *reinterpret_cast<uint4*>(sA[0] + lrow * SK + lseg + 8) = ga[1];
      const uint4* gu = reinterpret_cast<const uint4*>(
          upT + (size_t)(n0 + lrow) * D_DIM + k0 + lseg);
      *reinterpret_cast<uint4*>(sU[0] + lrow * SK + lseg)     = gu[0];
      *reinterpret_cast<uint4*>(sU[0] + lrow * SK + lseg + 8) = gu[1];
      const uint4* gg = reinterpret_cast<const uint4*>(
          gateT + (size_t)(n0 + lrow) * D_DIM + k0 + lseg);
      *reinterpret_cast<uint4*>(sG[0] + lrow * SK + lseg)     = gg[0];
      *reinterpret_cast<uint4*>(sG[0] + lrow * SK + lseg + 8) = gg[1];
    }
    __syncthreads();
    gemm1_tile(sA[0], sU[0], sG[0], wm, wn, lane, accU, accG);
    __syncthreads();
  }
#endif

  // epilogue: h = up * gelu_exact(gate) -> bf16
#pragma unroll
  for (int mf = 0; mf < 2; mf++) {
#pragma unroll
    for (int nf = 0; nf < 4; nf++) {
      const int n = n0 + wn * 64 + nf * 16 + (lane & 15);
      const v8f cu = accU[mf][nf];
      const v8f cg = accG[mf][nf];
#pragma unroll
      for (int r = 0; r < 8; r++) {
        const int m = m0 + wm * 32 + mf * 16 + (lane >> 4) * 8 + r;
        const float g   = cg[r];
        const float gel = 0.5f * g * (1.0f + erff(g * 0.70710678118654752f));
        H[(size_t)m * I_DIM + n] = f2bf(cu[r] * gel);
      }
    }
  }
}

// ---------------- GEMM2: out (+)= w_tok * (H @ down) --------------
// H: [T][I] bf16   downT: [D][I] bf16   out: [T][D] fp32
__global__ __launch_bounds__(256) void gemm2_down(
    const unsigned short* __restrict__ Hbf,
    const unsigned short* __restrict__ downT,
    const float* __restrict__ combine,
    float* __restrict__ out, int eidx, int beta) {
  __shared__ unsigned short sA[2][BM * SK];
  __shared__ unsigned short sB[2][BN * SK];

  const int tid  = threadIdx.x;
  const int lane = tid & 31;
  const int wid  = tid >> 5;
  const int wm   = wid & 3;
  const int wn   = wid >> 2;
  const int m0   = blockIdx.x * BM;
  const int n0   = blockIdx.y * BN;

  v8f acc[2][4];
  const v8f vz = {0.f, 0.f, 0.f, 0.f, 0.f, 0.f, 0.f, 0.f};
#pragma unroll
  for (int i = 0; i < 2; i++)
#pragma unroll
    for (int j = 0; j < 4; j++) acc[i][j] = vz;

  const int NK = I_DIM / BK;

#if USE_TDM
  unsigned int oA[2], oB[2];
#pragma unroll
  for (int b = 0; b < 2; b++) {
    oA[b] = (unsigned int)(uintptr_t)(const void*)&sA[b][0];
    oB[b] = (unsigned int)(uintptr_t)(const void*)&sB[b][0];
  }
  if (wid == 0)      tdm_load_tile(Hbf   + (size_t)m0 * I_DIM, oA[0], T_DIM, I_DIM, BM);
  else if (wid == 1) tdm_load_tile(downT + (size_t)n0 * I_DIM, oB[0], D_DIM, I_DIM, BN);

  for (int kt = 0; kt < NK; kt++) {
    const int cur = kt & 1, nxt = cur ^ 1;
    if (kt + 1 < NK) {
      const int k1 = (kt + 1) * BK;
      if (wid == 0)      tdm_load_tile(Hbf   + (size_t)m0 * I_DIM + k1, oA[nxt], T_DIM, I_DIM, BM);
      else if (wid == 1) tdm_load_tile(downT + (size_t)n0 * I_DIM + k1, oB[nxt], D_DIM, I_DIM, BN);
      __builtin_amdgcn_s_wait_tensorcnt(1);
    } else {
      __builtin_amdgcn_s_wait_tensorcnt(0);
    }
    __syncthreads();
    gemm2_tile(sA[cur], sB[cur], wm, wn, lane, acc);
    __syncthreads();
  }
#else
  const int lrow = tid >> 1;
  const int lseg = (tid & 1) * 16;
  for (int kt = 0; kt < NK; kt++) {
    const int k0 = kt * BK;
    {
      const uint4* ga = reinterpret_cast<const uint4*>(
          Hbf + (size_t)(m0 + lrow) * I_DIM + k0 + lseg);
      *reinterpret_cast<uint4*>(sA[0] + lrow * SK + lseg)     = ga[0];
      *reinterpret_cast<uint4*>(sA[0] + lrow * SK + lseg + 8) = ga[1];
      const uint4* gb = reinterpret_cast<const uint4*>(
          downT + (size_t)(n0 + lrow) * I_DIM + k0 + lseg);
      *reinterpret_cast<uint4*>(sB[0] + lrow * SK + lseg)     = gb[0];
      *reinterpret_cast<uint4*>(sB[0] + lrow * SK + lseg + 8) = gb[1];
    }
    __syncthreads();
    gemm2_tile(sA[0], sB[0], wm, wn, lane, acc);
    __syncthreads();
  }
#endif

  // epilogue: per-token combine weight, accumulate into fp32 output
#pragma unroll
  for (int mf = 0; mf < 2; mf++) {
#pragma unroll
    for (int nf = 0; nf < 4; nf++) {
      const int n = n0 + wn * 64 + nf * 16 + (lane & 15);
      const v8f c = acc[mf][nf];
#pragma unroll
      for (int r = 0; r < 8; r++) {
        const int m = m0 + wm * 32 + mf * 16 + (lane >> 4) * 8 + r;
        const float w = (eidx >= 0) ? combine[(size_t)m * E_DIM + eidx] : 1.0f;
        const size_t o = (size_t)m * D_DIM + n;
        const float prev = beta ? out[o] : 0.0f;
        out[o] = prev + w * c[r];
      }
    }
  }
}

// ------------------------------------------------------------------
extern "C" void kernel_launch(void* const* d_in, const int* in_sizes, int n_in,
                              void* d_out, int out_size, void* d_ws, size_t ws_size,
                              hipStream_t stream) {
  (void)in_sizes; (void)n_in; (void)out_size; (void)ws_size;

  const float* x           = (const float*)d_in[0];
  const float* gate_w      = (const float*)d_in[1];
  const float* up_w        = (const float*)d_in[2];
  const float* gproj_w     = (const float*)d_in[3];
  const float* down_w      = (const float*)d_in[4];
  const float* shared_up   = (const float*)d_in[5];
  const float* shared_gate = (const float*)d_in[6];
  const float* shared_down = (const float*)d_in[7];
  float* out = (float*)d_out;

  // workspace layout (bytes)
  char* ws = (char*)d_ws;
  const size_t OFF_COMBINE = 0;                                     // 128 KB
  const size_t OFF_XBF     = OFF_COMBINE + 131072;                  // 8 MB
  const size_t OFF_UPT     = OFF_XBF + (size_t)T_DIM * D_DIM * 2;   // 1 MB
  const size_t OFF_GATET   = OFF_UPT + (size_t)I_DIM * D_DIM * 2;   // 1 MB
  const size_t OFF_DOWNT   = OFF_GATET + (size_t)I_DIM * D_DIM * 2; // 1 MB
  const size_t OFF_H       = OFF_DOWNT + (size_t)D_DIM * I_DIM * 2; // 4 MB

  float*          combine = (float*)(ws + OFF_COMBINE);
  unsigned short* xbf     = (unsigned short*)(ws + OFF_XBF);
  unsigned short* upT     = (unsigned short*)(ws + OFF_UPT);
  unsigned short* gateT   = (unsigned short*)(ws + OFF_GATET);
  unsigned short* downT   = (unsigned short*)(ws + OFF_DOWNT);
  unsigned short* Hbf     = (unsigned short*)(ws + OFF_H);

  // 1) router -> dense combine weights
  router_kernel<<<T_DIM / 8, 256, 0, stream>>>(x, gate_w, combine);

  // 2) activations -> bf16
  cvt_kernel<<<2048, 256, 0, stream>>>(x, xbf, T_DIM * D_DIM);

  // 3) experts, serialized on the stream (weight buffers + H reused safely).
  //    shared expert first initializes the output (beta=0).
  auto run_expert = [&](const float* uw, const float* gw, const float* dw,
                        int eidx, int beta) {
    cvtT_kernel<<<1024, 256, 0, stream>>>(uw, upT,   D_DIM, I_DIM); // [D][I] -> [I][D]
    cvtT_kernel<<<1024, 256, 0, stream>>>(gw, gateT, D_DIM, I_DIM);
    cvtT_kernel<<<1024, 256, 0, stream>>>(dw, downT, I_DIM, D_DIM); // [I][D] -> [D][I]
    gemm1_upgate<<<dim3(T_DIM / BM, I_DIM / BN), 256, 0, stream>>>(
        xbf, upT, gateT, Hbf);
    gemm2_down<<<dim3(T_DIM / BM, D_DIM / BN), 256, 0, stream>>>(
        Hbf, downT, combine, out, eidx, beta);
  };

  run_expert(shared_up, shared_gate, shared_down, /*eidx=*/-1, /*beta=*/0);
  for (int e = 0; e < E_DIM; e++) {
    run_expert(up_w    + (size_t)e * D_DIM * I_DIM,
               gproj_w + (size_t)e * D_DIM * I_DIM,
               down_w  + (size_t)e * I_DIM * D_DIM,
               e, /*beta=*/1);
  }
}
